// SP_CAM_withDCR_52682068852848
// MI455X (gfx1250) — matrix operand order repaired and checked by reading, not compiled, not run on hardware
//
#include <hip/hip_runtime.h>
#include <hip/hip_bf16.h>
#include <math.h>

// ---------------------------------------------------------------------------
// Types for CDNA5 WMMA (wave32): v_wmma_f32_16x16x32_bf16
// ---------------------------------------------------------------------------
typedef __attribute__((ext_vector_type(16))) __bf16 v16bf;
typedef __attribute__((ext_vector_type(8)))  float  v8f;

#define BM 128  // block tile M (out channels)
#define BN 64   // block tile N (output pixels)
#define BK 32   // K step == WMMA K for bf16

// Async global->LDS path (GLOBAL_LOAD_ASYNC_TO_LDS_B128, ASYNCcnt).
// Probe-discovered signature: (v4i addrspace(1)*, v4i addrspace(3)*, imm, imm)
#if defined(__HIP_DEVICE_COMPILE__) && \
    __has_builtin(__builtin_amdgcn_global_load_async_to_lds_b128) && \
    __has_builtin(__builtin_amdgcn_s_wait_asynccnt)
#define USE_ASYNC_LDS 1
typedef __attribute__((ext_vector_type(4))) int v4i_a;
typedef __attribute__((address_space(1))) v4i_a* as1p;
typedef __attribute__((address_space(3))) v4i_a* as3p;
#else
#define USE_ASYNC_LDS 0
#endif

__device__ __forceinline__ unsigned short f32_bf16(float f) {
  unsigned int u = __float_as_uint(f);
  u += 0x7fffu + ((u >> 16) & 1u);   // round to nearest even
  return (unsigned short)(u >> 16);
}

// Convert fp32 weights [O][K] -> bf16 [O][Kp] with zero K-padding so the
// A-tile async copies are always aligned, in-bounds, and zero-padded.
__global__ void f32_to_bf16_pad_k(const float* __restrict__ x,
                                  unsigned short* __restrict__ y,
                                  int O, int K, int Kp) {
  int i = blockIdx.x * blockDim.x + threadIdx.x;
  int total = O * Kp;
  if (i >= total) return;
  int o = i / Kp, k = i - o * Kp;
  y[i] = (k < K) ? f32_bf16(x[o * K + k]) : (unsigned short)0;
}

// ---------------------------------------------------------------------------
// Implicit-GEMM convolution, bf16 WMMA, fused scale/bias/residual/activation.
//   A = weights [O][Kp] (bf16, zero K-pad), B = im2col(X), N = Bn*OH*OW
//   256 threads = 8 waves as 4(M) x 2(N); each wave owns a 32x32 macro-tile
//   -> 4 v_wmma_f32_16x16x32_bf16 per wave per K-chunk. Double-buffered LDS;
//   A tile streamed with global_load_async_to_lds_b128 when available.
// ---------------------------------------------------------------------------
template <int TKH, int TKW>
__global__ __launch_bounds__(256)
void conv_wmma_k(const unsigned short* __restrict__ Wb,
                 const float* __restrict__ X,
                 float* __restrict__ Y,
                 const float* __restrict__ scale,
                 const float* __restrict__ bias,
                 const float* __restrict__ resid,
                 int Bn, int C, int H, int W,
                 int O, int OH, int OW,
                 int KH, int KW, int stride, int pad,
                 int act, int K, int Kp, int N) {
  __shared__ unsigned short As[2][BM][BK];   // 16 KB
  __shared__ unsigned short Bs[2][BN][BK];   //  8 KB

  const int tid  = threadIdx.x;
  const int lane = tid & 31;
  const int wave = tid >> 5;
  const int wm   = wave >> 1;          // 0..3 : M macro-tile (32 rows)
  const int wn   = wave & 1;           // 0..1 : N macro-tile (32 cols)
  const int m0   = blockIdx.y * BM;
  const int n0   = blockIdx.x * BN;
  const int KHW  = (TKH > 0) ? (TKH * TKW) : (KH * KW);
  const int OHW  = OH * OW;

  v8f acc[2][2];
  #pragma unroll
  for (int mi = 0; mi < 2; ++mi)
    #pragma unroll
    for (int ni = 0; ni < 2; ++ni)
      acc[mi][ni] = (v8f){0.f, 0.f, 0.f, 0.f, 0.f, 0.f, 0.f, 0.f};

  const int nch = Kp / BK;             // Kp is a multiple of 32

  // A-fill: 128x32 halfs / 256 thr = 16 contiguous halfs (32 B) each
  const int ar  = tid >> 1;            // row 0..127
  const int akq = (tid & 1) * 16;      // k-sub 0/16
  // clamp M row so async reads stay inside the padded weight buffer
  const int gmc = (m0 + ar < O) ? (m0 + ar) : (O - 1);
  const unsigned short* arow = Wb + (size_t)gmc * Kp + akq;

  // B-fill: 64x32 halfs / 256 thr = 8 halfs each
  const int bn_ = tid >> 2;            // pixel 0..63
  const int bkq = (tid & 3) * 8;       // k-sub 0..24

  const int gn = n0 + bn_;
  int pb = 0, poy = 0, pox = 0;
  if (gn < N) {
    pb = gn / OHW;
    int p = gn - pb * OHW;
    poy = p / OW;
    pox = p - poy * OW;
  }

  auto fill = [&](int kc, int buf) {
    const int k0 = kc * BK;
    // ---- A tile: async DMA global->LDS (or scalar fallback) ----
    {
      const unsigned short* src = arow + k0;
#if USE_ASYNC_LDS
      __builtin_amdgcn_global_load_async_to_lds_b128(
          (as1p)src, (as3p)&As[buf][ar][akq], 0, 0);
      __builtin_amdgcn_global_load_async_to_lds_b128(
          (as1p)(src + 8), (as3p)&As[buf][ar][akq + 8], 0, 0);
#else
      #pragma unroll
      for (int j = 0; j < 16; ++j) As[buf][ar][akq + j] = src[j];
#endif
      // speculative prefetch of the tile after next (global_prefetch_b8)
      __builtin_prefetch(src + 2 * BK, 0, 0);
    }
    // ---- B tile (on-the-fly im2col, fp32 -> bf16) ----
    {
      #pragma unroll
      for (int j = 0; j < 8; ++j) {
        int kk = k0 + bkq + j;
        float v = 0.f;
        if (gn < N && kk < K) {
          int c, ky, kx;
          if (TKH == 1 && TKW == 1) {
            c = kk; ky = 0; kx = 0;
          } else if (TKH > 0) {
            // divisors clamped to >=1 so the dead generic instantiation
            // does not trip -Wdivision-by-zero; folds to constants here
            const int khw_c = (TKH * TKW > 0) ? (TKH * TKW) : 1;
            const int kw_c  = (TKW > 0) ? TKW : 1;
            c = kk / khw_c;
            int r = kk - c * khw_c;
            ky = r / kw_c;
            kx = r - ky * kw_c;
          } else {
            c = kk / KHW;
            int r = kk - c * KHW;
            ky = r / KW;
            kx = r - ky * KW;
          }
          int iy = poy * stride - pad + ky;
          int ix = pox * stride - pad + kx;
          if (iy >= 0 && iy < H && ix >= 0 && ix < W)
            v = X[((pb * C + c) * H + iy) * W + ix];
        }
        Bs[buf][bn_][bkq + j] = f32_bf16(v);
      }
    }
  };

  fill(0, 0);
#if USE_ASYNC_LDS
  __builtin_amdgcn_s_wait_asynccnt(0);
#endif
  __syncthreads();

  for (int kc = 0; kc < nch; ++kc) {
    const int buf = kc & 1;
    if (kc + 1 < nch) fill(kc + 1, buf ^ 1);

    // A fragments: ISA 16-bit A 16x32 layout per 16-row sub-tile
    //   lanes 0-15 (M=lane): K 0..7 and 16..23 ; lanes 16-31: K 8..15, 24..31
    union Frag { v16bf v; unsigned short s[16]; };
    Frag a[2], b[2];
    {
      const int kh8 = (lane >> 4) * 8;
      #pragma unroll
      for (int mi = 0; mi < 2; ++mi) {
        const unsigned short* ap =
            &As[buf][wm * 32 + mi * 16 + (lane & 15)][0];
        #pragma unroll
        for (int j = 0; j < 8; ++j) {
          a[mi].s[j]     = ap[kh8 + j];
          a[mi].s[8 + j] = ap[kh8 + 16 + j];
        }
      }
      // B fragments: same K order as A (lanes 0-15: K 0..7,16..23; 16-31: K 8..15,24..31)
      #pragma unroll
      for (int ni = 0; ni < 2; ++ni) {
        const unsigned short* bp =
            &Bs[buf][wn * 32 + ni * 16 + (lane & 15)][(lane >> 4) * 8];
        #pragma unroll
        for (int j = 0; j < 8; ++j) {
          b[ni].s[j]     = bp[j];
          b[ni].s[8 + j] = bp[16 + j];
        }
      }
    }

    #pragma unroll
    for (int mi = 0; mi < 2; ++mi)
      #pragma unroll
      for (int ni = 0; ni < 2; ++ni)
        acc[mi][ni] = __builtin_amdgcn_wmma_f32_16x16x32_bf16(
            false, a[mi].v, false, b[ni].v, (short)0, acc[mi][ni],
            false, false);

#if USE_ASYNC_LDS
    __builtin_amdgcn_s_wait_asynccnt(0);
#endif
    __syncthreads();
  }

  // Epilogue: C/D layout — lane<16: M = v ; lane>=16: M = v + 8 ; N = lane&15
  #pragma unroll
  for (int ni = 0; ni < 2; ++ni) {
    const int gn_out = n0 + wn * 32 + ni * 16 + (lane & 15);
    if (gn_out >= N) continue;
    int ob = gn_out / OHW;
    int op = gn_out - ob * OHW;
    #pragma unroll
    for (int mi = 0; mi < 2; ++mi) {
      const int mbase = m0 + wm * 32 + mi * 16 + (lane >> 4) * 8;
      #pragma unroll
      for (int v = 0; v < 8; ++v) {
        int gm = mbase + v;
        if (gm < O) {
          float s   = scale ? scale[gm] : 1.f;
          float bsv = bias ? bias[gm] : 0.f;
          float val = acc[mi][ni][v] * s + bsv;
          int idx = (ob * O + gm) * OHW + op;
          if (resid) val += resid[idx];
          if (act == 1)       val = fmaxf(val, 0.f);
          else if (act == 2)  val = (val > 0.f) ? val : 0.1f * val;
          Y[idx] = val;
        }
      }
    }
  }
}

// ---------------------------------------------------------------------------
// Elementwise / reduction glue kernels
// ---------------------------------------------------------------------------
__global__ void maxpool_k(const float* __restrict__ x, float* __restrict__ y,
                          int Bn, int C, int H, int W, int OH, int OW) {
  int i = blockIdx.x * blockDim.x + threadIdx.x;
  int total = Bn * C * OH * OW;
  if (i >= total) return;
  int ox = i % OW; int t = i / OW;
  int oy = t % OH; t /= OH;
  int c = t % C;   int b = t / C;
  float m = -INFINITY;
  for (int dy = 0; dy < 3; ++dy)
    for (int dx = 0; dx < 3; ++dx) {
      int iy = oy * 2 - 1 + dy, ix = ox * 2 - 1 + dx;
      if (iy >= 0 && iy < H && ix >= 0 && ix < W)
        m = fmaxf(m, x[((b * C + c) * H + iy) * W + ix]);
    }
  y[i] = m;
}

__global__ void softmax9_k(float* __restrict__ x, int Bn, int HW) {
  int i = blockIdx.x * blockDim.x + threadIdx.x;
  if (i >= Bn * HW) return;
  int b = i / HW, p = i - b * HW;
  float v[9], mx = -INFINITY;
  for (int k = 0; k < 9; ++k) { v[k] = x[(b * 9 + k) * HW + p]; mx = fmaxf(mx, v[k]); }
  float s = 0.f;
  for (int k = 0; k < 9; ++k) { v[k] = expf(v[k] - mx); s += v[k]; }
  float inv = 1.f / s;
  for (int k = 0; k < 9; ++k) x[(b * 9 + k) * HW + p] = v[k] * inv;
}

// out[b,c,y,x] = ala[b,c] * sum_k x[b,c,y+dy-1,x+dx-1] * prob[b,k,y,x]
__global__ void upfeat_k(const float* __restrict__ x, const float* __restrict__ prob,
                         const float* __restrict__ ala, float* __restrict__ y,
                         int Bn, int C, int H, int W) {
  int i = blockIdx.x * blockDim.x + threadIdx.x;
  int HW = H * W;
  int total = Bn * C * HW;
  if (i >= total) return;
  int xx = i % W; int t = i / W;
  int yy = t % H; t /= H;
  int c = t % C;  int b = t / C;
  float s = ala ? ala[b * C + c] : 1.f;
  const float* pb = prob + (size_t)b * 9 * HW;
  const float* xb = x + (size_t)((b * C + c) * H) * W;
  float acc = 0.f;
  int k = 0;
  for (int dy = 0; dy < 3; ++dy)
    for (int dx = 0; dx < 3; ++dx, ++k) {
      int iy = yy + dy - 1, ix = xx + dx - 1;
      if (iy >= 0 && iy < H && ix >= 0 && ix < W)
        acc += xb[iy * W + ix] * pb[k * HW + yy * W + xx];
    }
  y[i] = acc * s;
}

__global__ void gap_k(const float* __restrict__ x, float* __restrict__ g,
                      int Bn, int C, int HW) {
  int i = blockIdx.x * blockDim.x + threadIdx.x;
  if (i >= Bn * C) return;
  const float* p = x + (size_t)i * HW;
  float s = 0.f;
  for (int j = 0; j < HW; ++j) s += p[j];
  g[i] = s / (float)HW;
}

// y[b,o] = act( sum_c w[o,c] * x[b,c] ) ; act: 1=relu, 3=sigmoid
__global__ void gemv_k(const float* __restrict__ w, const float* __restrict__ x,
                       float* __restrict__ y, int Bn, int O, int C, int act) {
  int i = blockIdx.x * blockDim.x + threadIdx.x;
  if (i >= Bn * O) return;
  int b = i / O, o = i - b * O;
  const float* wr = w + (size_t)o * C;
  const float* xr = x + (size_t)b * C;
  float s = 0.f;
  for (int c = 0; c < C; ++c) s += wr[c] * xr[c];
  if (act == 1)      s = fmaxf(s, 0.f);
  else if (act == 3) s = 1.f / (1.f + expf(-s));
  y[i] = s;
}

__global__ void scale_ch_k(const float* __restrict__ x, const float* __restrict__ ala,
                           float* __restrict__ y, int Bn, int C, int HW) {
  int i = blockIdx.x * blockDim.x + threadIdx.x;
  int total = Bn * C * HW;
  if (i >= total) return;
  int bc = i / HW;
  y[i] = x[i] * ala[bc];
}

__global__ void meanhw_k(const float* __restrict__ x, float* __restrict__ y,
                         int Bn, int C, int HW) {
  int i = blockIdx.x * blockDim.x + threadIdx.x;
  if (i >= Bn * C) return;
  const float* p = x + (size_t)i * HW;
  float s = 0.f;
  for (int j = 0; j < HW; ++j) s += p[j];
  y[i] = s / (float)HW;
}

// ---------------------------------------------------------------------------
// Host orchestration
// ---------------------------------------------------------------------------
extern "C" void kernel_launch(void* const* d_in, const int* in_sizes, int n_in,
                              void* d_out, int out_size, void* d_ws, size_t ws_size,
                              hipStream_t stream) {
  (void)in_sizes; (void)n_in; (void)out_size; (void)ws_size;

  const float* inputs = (const float*)d_in[0];
  // d_in[1] = probs feeds only the dead `qf` branch (deleted in reference) — skip.

  int ci = 2;  // params flattened in JAX sorted-key order
  auto nxt = [&]() -> const float* { return (const float*)d_in[ci++]; };

  const float* ala1_w1 = nxt(); const float* ala1_w2 = nxt();
  const float* ala2_w1 = nxt(); const float* ala2_w2 = nxt();
  const float* bn1_b   = nxt(); const float* bn1_s   = nxt();
  const float* cls_w   = nxt();
  const float* conv1_w = nxt();

  struct BnP { const float *b, *s; };
  struct Blk { BnP bn1, bn2, bn3, bnd; const float *w1, *w2, *w3, *wd; int has_d; };
  Blk blocks[16]; int nb = 0;
  const int planes[4]  = {64, 128, 256, 512};
  const int nblk[4]    = {3, 4, 6, 3};
  const int lstride[4] = {1, 2, 2, 1};
  for (int L = 0; L < 4; ++L)
    for (int b = 0; b < nblk[L]; ++b) {
      Blk bl{}; bool hd = (b == 0);
      bl.bn1.b = nxt(); bl.bn1.s = nxt();
      bl.bn2.b = nxt(); bl.bn2.s = nxt();
      bl.bn3.b = nxt(); bl.bn3.s = nxt();
      if (hd) { bl.bnd.b = nxt(); bl.bnd.s = nxt(); }
      bl.w1 = nxt(); bl.w2 = nxt(); bl.w3 = nxt();
      if (hd) bl.wd = nxt();
      bl.has_d = hd; blocks[nb++] = bl;
    }
  ci += 8;  // qf: b3, bn1.b, bn1.s, bn2.b, bn2.s, w1, w2, w3 — dead branch
  const float* t4_b1 = nxt(); const float* t4_b2 = nxt();
  const float* t4_w1 = nxt(); const float* t4_w2 = nxt();
  const float* t5_b1 = nxt(); const float* t5_b2 = nxt();
  const float* t5_w1 = nxt(); const float* t5_w2 = nxt();
  const float* x4f_b = nxt(); const float* x4f_s = nxt(); const float* x4f_w = nxt();
  const float* x5f_b = nxt(); const float* x5f_s = nxt(); const float* x5f_w = nxt();

  // ---- workspace bump allocator ----
  char* wsp = (char*)d_ws; size_t off = 0;
  auto alloc = [&](size_t bytes) -> void* {
    off = (off + 255) & ~(size_t)255;
    void* p = wsp + off; off += bytes; return p;
  };
  unsigned short* wtmp = (unsigned short*)alloc((size_t)8  << 20);  // bf16 weights
  float* bufX  = (float*)alloc((size_t)68 << 20);
  float* bufY  = (float*)alloc((size_t)68 << 20);
  float* bufS  = (float*)alloc((size_t)68 << 20);
  float* bufT1 = (float*)alloc((size_t)20 << 20);
  float* bufT2 = (float*)alloc((size_t)20 << 20);
  float* dpbuf = (float*)alloc((size_t)1  << 20);
  float* gapbuf = (float*)alloc((size_t)64 << 10);
  float* hidbuf = (float*)alloc((size_t)64 << 10);
  float* alabuf = (float*)alloc((size_t)64 << 10);
  float* clsbuf = (float*)alloc((size_t)1  << 20);

  auto conv = [&](const float* w, int O, int C, int KH, int KW,
                  const float* x, int Bn, int H, int W, int stride,
                  const float* scale, const float* bias, const float* resid,
                  int act, float* y, int* pOH, int* pOW) {
    int pad = (KH - 1) / 2;
    int OH = (H + 2 * pad - KH) / stride + 1;
    int OW = (W + 2 * pad - KW) / stride + 1;
    int K = C * KH * KW, N = Bn * OH * OW;
    int Kp = (K + 31) & ~31;  // zero-padded K so A tiles are aligned/full
    int nw = O * Kp;
    f32_to_bf16_pad_k<<<(nw + 255) / 256, 256, 0, stream>>>(w, wtmp, O, K, Kp);
    dim3 grid((N + BN - 1) / BN, (O + BM - 1) / BM);
    // constant-divisor specializations for every window in this network
    auto kern = (KH == 1 && KW == 1) ? conv_wmma_k<1, 1>
              : (KH == 3 && KW == 3) ? conv_wmma_k<3, 3>
              : (KH == 7 && KW == 7) ? conv_wmma_k<7, 7>
                                     : conv_wmma_k<0, 0>;
    kern<<<grid, 256, 0, stream>>>(wtmp, x, y, scale, bias, resid,
                                   Bn, C, H, W, O, OH, OW,
                                   KH, KW, stride, pad, act, K, Kp, N);
    if (pOH) *pOH = OH;
    if (pOW) *pOW = OW;
  };

  auto bottleneck = [&](const Blk& bl, int pl, int s, const float* x,
                        int Cin, int H, int W, float* y, int* pH, int* pW) {
    int oh, ow, oh2, ow2;
    conv(bl.w1, pl, Cin, 1, 1, x, 4, H, W, 1,
         bl.bn1.s, bl.bn1.b, nullptr, 1, bufT1, &oh, &ow);
    conv(bl.w2, pl, pl, 3, 3, bufT1, 4, H, W, s,
         bl.bn2.s, bl.bn2.b, nullptr, 1, bufT2, &oh2, &ow2);
    const float* res = x;
    if (bl.has_d) {
      conv(bl.wd, pl * 4, Cin, 1, 1, x, 4, H, W, s,
           bl.bnd.s, bl.bnd.b, nullptr, 0, bufS, nullptr, nullptr);
      res = bufS;
    }
    conv(bl.w3, pl * 4, pl, 1, 1, bufT2, 4, oh2, ow2, 1,
         bl.bn3.s, bl.bn3.b, res, 1, y, nullptr, nullptr);
    *pH = oh2; *pW = ow2;
  };

  // ---- stem ----
  int H, W, oh, ow;
  conv(conv1_w, 64, 3, 7, 7, inputs, 4, 512, 512, 2,
       bn1_s, bn1_b, nullptr, 1, bufY, &oh, &ow);           // [4,64,256,256]
  int ph = (oh + 2 - 3) / 2 + 1, pw = (ow + 2 - 3) / 2 + 1; // 128x128
  {
    int total = 4 * 64 * ph * pw;
    maxpool_k<<<(total + 255) / 256, 256, 0, stream>>>(bufY, bufX, 4, 64, oh, ow, ph, pw);
  }
  float* x = bufX; float* y = bufY;
  int Cin = 64; H = ph; W = pw;

  // ---- layers 0..2 ----
  int bi = 0;
  for (int L = 0; L < 3; ++L)
    for (int b = 0; b < nblk[L]; ++b) {
      int s = (b == 0) ? lstride[L] : 1;
      int nh, nw2;
      bottleneck(blocks[bi++], planes[L], s, x, Cin, H, W, y, &nh, &nw2);
      Cin = planes[L] * 4; H = nh; W = nw2;
      float* t = x; x = y; y = t;
    }
  // x = x4 [4,1024,32,32]
  int HW = H * W;

  // ---- heads on x4 ----
  conv(x4f_w, 128, 1024, 1, 1, x, 4, H, W, 1, x4f_s, x4f_b, nullptr, 2, bufT1, nullptr, nullptr);
  conv(t4_w1, 256, 128, 3, 3, bufT1, 4, H, W, 1, nullptr, t4_b1, nullptr, 2, bufT2, nullptr, nullptr);
  conv(t4_w2, 9, 256, 1, 1, bufT2, 4, H, W, 1, nullptr, t4_b2, nullptr, 2, dpbuf, nullptr, nullptr);
  softmax9_k<<<(4 * HW + 255) / 256, 256, 0, stream>>>(dpbuf, 4, HW);
  gap_k<<<(4 * 1024 + 255) / 256, 256, 0, stream>>>(x, gapbuf, 4, 1024, HW);
  gemv_k<<<(4 * 64 + 63) / 64, 64, 0, stream>>>(ala1_w1, gapbuf, hidbuf, 4, 64, 1024, 1);
  gemv_k<<<(4 * 1024 + 255) / 256, 256, 0, stream>>>(ala1_w2, hidbuf, alabuf, 4, 1024, 64, 3);
  {
    int total = 4 * 1024 * HW;
    upfeat_k<<<(total + 255) / 256, 256, 0, stream>>>(x, dpbuf, alabuf, y, 4, 1024, H, W);
  }
  { float* t = x; x = y; y = t; }

  // ---- layer 3 ----
  for (int b = 0; b < nblk[3]; ++b) {
    int s = (b == 0) ? lstride[3] : 1;
    int nh, nw2;
    bottleneck(blocks[bi++], planes[3], s, x, Cin, H, W, y, &nh, &nw2);
    Cin = planes[3] * 4; H = nh; W = nw2;
    float* t = x; x = y; y = t;
  }
  HW = H * W;  // x = x5 [4,2048,32,32]

  // ---- heads on x5 ----
  conv(x5f_w, 128, 2048, 1, 1, x, 4, H, W, 1, x5f_s, x5f_b, nullptr, 2, bufT1, nullptr, nullptr);
  conv(t5_w1, 256, 128, 3, 3, bufT1, 4, H, W, 1, nullptr, t5_b1, nullptr, 2, bufT2, nullptr, nullptr);
  conv(t5_w2, 9, 256, 1, 1, bufT2, 4, H, W, 1, nullptr, t5_b2, nullptr, 2, dpbuf, nullptr, nullptr);
  softmax9_k<<<(4 * HW + 255) / 256, 256, 0, stream>>>(dpbuf, 4, HW);
  gap_k<<<(4 * 2048 + 255) / 256, 256, 0, stream>>>(x, gapbuf, 4, 2048, HW);
  gemv_k<<<(4 * 128 + 127) / 128, 128, 0, stream>>>(ala2_w1, gapbuf, hidbuf, 4, 128, 2048, 1);
  gemv_k<<<(4 * 2048 + 255) / 256, 256, 0, stream>>>(ala2_w2, hidbuf, alabuf, 4, 2048, 128, 3);
  {
    int total = 4 * 2048 * HW;
    scale_ch_k<<<(total + 255) / 256, 256, 0, stream>>>(x, alabuf, y, 4, 2048, HW);
  }
  { float* t = x; x = y; y = t; }

  // ---- classifier + outputs ----
  conv(cls_w, 21, 2048, 1, 1, x, 4, H, W, 1, nullptr, nullptr, nullptr, 0,
       clsbuf, nullptr, nullptr);
  float* out = (float*)d_out;
  meanhw_k<<<(4 * 21 + 63) / 64, 64, 0, stream>>>(clsbuf, out + 4 * 21 * HW, 4, 21, HW);
  {
    int total = 4 * 21 * HW;
    upfeat_k<<<(total + 255) / 256, 256, 0, stream>>>(clsbuf, dpbuf, nullptr, out, 4, 21, H, W);
  }
}
